// RNN_11673721111233
// MI455X (gfx1250) — matrix operand order repaired and checked by reading, not compiled, not run on hardware
//
#include <hip/hip_runtime.h>
#include <hip/hip_bf16.h>

typedef __attribute__((ext_vector_type(16))) _Float16 v16h;
typedef __attribute__((ext_vector_type(8)))  _Float16 v8h;
typedef __attribute__((ext_vector_type(8)))  float    v8f;

#define B_    16384
#define T_    32
#define V_    27
#define E_    64
#define H_    64
#define K_    128              // E + H
#define KPAD  136              // padded halves per xh / weight row (272B: bank-conflict-free)
#define WLMP  72               // padded halves per Wlm row
#define WAVES 8
#define RPW   16               // batch rows per wave (WMMA M)

#define WMMA_F16(a, b, c) \
  __builtin_amdgcn_wmma_f32_16x16x32_f16(false, (a), false, (b), (short)0, (c), false, false)

__device__ __forceinline__ float sigmoidf_(float x) { return 1.0f / (1.0f + __expf(-x)); }
__device__ __forceinline__ float tanhf_(float x)    { return 1.0f - 2.0f / (__expf(2.0f * x) + 1.0f); }

__global__ __launch_bounds__(32 * WAVES) void gru_wmma_kernel(
    const int*   __restrict__ idx,
    const float* __restrict__ wte,
    const float* __restrict__ start,
    const float* __restrict__ Wz, const float* __restrict__ bz,
    const float* __restrict__ Wr, const float* __restrict__ br,
    const float* __restrict__ Wh, const float* __restrict__ bh,
    const float* __restrict__ Wlm, const float* __restrict__ blm,
    float* __restrict__ out)
{
  __shared__ _Float16 sWz[H_ * KPAD];
  __shared__ _Float16 sWr[H_ * KPAD];
  __shared__ _Float16 sWh[H_ * KPAD];
  __shared__ _Float16 sWlm[32 * WLMP];
  __shared__ _Float16 sWte[V_ * E_];
  __shared__ _Float16 sXH[WAVES][RPW * KPAD];
  __shared__ int      sIdx[WAVES][RPW];

  const int tid = threadIdx.x;

  // ---- cooperative f32 -> f16 weight staging into LDS ----
  for (int i = tid; i < H_ * K_; i += 32 * WAVES) {
    int n = i >> 7, k = i & 127;
    sWz[n * KPAD + k] = (_Float16)Wz[i];
    sWr[n * KPAD + k] = (_Float16)Wr[i];
    sWh[n * KPAD + k] = (_Float16)Wh[i];
  }
  for (int i = tid; i < 32 * H_; i += 32 * WAVES) {
    int n = i >> 6, k = i & 63;
    float v = (n < V_) ? Wlm[n * H_ + k] : 0.0f;
    sWlm[n * WLMP + k] = (_Float16)v;
  }
  for (int i = tid; i < V_ * E_; i += 32 * WAVES)
    sWte[i] = (_Float16)wte[i];
  __syncthreads();

  const int wave = tid >> 5;
  const int lane = tid & 31;
  const int hi   = (lane >> 4) & 1;  // which half-wave
  const int lm   = lane & 15;        // A row / B,C,D column within 16-tile
  const int mb   = hi ? 8 : 0;       // C/D row base
  const int row0 = (blockIdx.x * WAVES + wave) * RPW;

  _Float16* xh = sXH[wave];
  const _Float16* arow = xh + lm * KPAD;

  // ---- init hidden state h0 = start (broadcast), vectorized 8-half chunks ----
  for (int c = lane; c < RPW * 8; c += 32) {
    int m = c >> 3, k = (c & 7) * 8;
    v8h v;
#pragma unroll
    for (int i = 0; i < 8; ++i) v[i] = (_Float16)start[k + i];
    *(v8h*)&xh[m * KPAD + E_ + k] = v;
  }

  // ---- per-lane hoisted column indices & biases ----
  int   ncol[4];  float bzv[4], brv[4], bhv[4];
#pragma unroll
  for (int nt = 0; nt < 4; ++nt) {
    ncol[nt] = nt * 16 + lm;
    bzv[nt] = bz[ncol[nt]];
    brv[nt] = br[ncol[nt]];
    bhv[nt] = bh[ncol[nt]];
  }
  int vcol[2]; float blmv[2];
#pragma unroll
  for (int vt = 0; vt < 2; ++vt) {
    vcol[vt] = vt * 16 + lm;
    blmv[vt] = (vcol[vt] < V_) ? blm[vcol[vt]] : 0.0f;
  }

  // A fragment: 16x32 f16 tile, rows = lm; lane holds K = cb..cb+7 and cb+16..cb+23
  auto loadA = [&](int kc) -> v16h {
    const _Float16* p = arow + kc * 32 + hi * 8;
    v8h lo = *(const v8h*)(p);
    v8h hh = *(const v8h*)(p + 16);
    v16h a;
#pragma unroll
    for (int i = 0; i < 8; ++i) { a[i] = lo[i]; a[i + 8] = hh[i]; }
    return a;
  };
  // B fragment: 32x16 f16; lane holds col n, K = cb..cb+7 and cb+16..cb+23 (cb = kc*32 + 8*hi)
  auto loadB = [&](const _Float16* w, int stride, int n, int kc) -> v16h {
    const _Float16* p = w + n * stride + kc * 32 + hi * 8;
    v8h lo = *(const v8h*)(p);
    v8h hh = *(const v8h*)(p + 16);
    v16h b;
#pragma unroll
    for (int i = 0; i < 8; ++i) { b[i] = lo[i]; b[i + 8] = hh[i]; }
    return b;
  };

  // ---- loop-invariant LM-head B fragments held in registers ----
  v16h wlmFrag[2][2];
#pragma unroll
  for (int vt = 0; vt < 2; ++vt)
#pragma unroll
    for (int kc = 0; kc < 2; ++kc)
      wlmFrag[vt][kc] = loadB(sWlm, WLMP, vcol[vt], kc);

  for (int t = 0; t < T_; ++t) {
    // stage this step's token indices (lanes 0..15), then copy embeddings in
    // 16-byte chunks: 128 chunks of 8 halves across the wave
    if (lane < RPW) sIdx[wave][lane] = idx[(row0 + lane) * T_ + t];
#pragma unroll
    for (int c0 = 0; c0 < 4; ++c0) {
      int c = c0 * 32 + lane;
      int m = c >> 3, e = (c & 7) * 8;
      int ix = sIdx[wave][m];
      *(v8h*)&xh[m * KPAD + e] = *(const v8h*)&sWte[ix * E_ + e];
    }

    // A fragments over full K = [x | h]
    v16h afr[4];
#pragma unroll
    for (int kc = 0; kc < 4; ++kc) afr[kc] = loadA(kc);

    // ---- gates z, r ----
    v8f zf[4];
    float hcur[4][8];
    float rv[4][8];
#pragma unroll
    for (int nt = 0; nt < 4; ++nt) {
      v8f accz = {}; v8f accr = {};
#pragma unroll
      for (int kc = 0; kc < 4; ++kc) {
        accz = WMMA_F16(afr[kc], loadB(sWz, KPAD, ncol[nt], kc), accz);
        accr = WMMA_F16(afr[kc], loadB(sWr, KPAD, ncol[nt], kc), accr);
      }
#pragma unroll
      for (int r = 0; r < 8; ++r) {
        zf[nt][r] = sigmoidf_(accz[r] + bzv[nt]);
        rv[nt][r] = sigmoidf_(accr[r] + brv[nt]);
      }
    }

    // capture h (D-layout addressing), then overwrite h-region with r*h
#pragma unroll
    for (int nt = 0; nt < 4; ++nt)
#pragma unroll
      for (int r = 0; r < 8; ++r)
        hcur[nt][r] = (float)xh[(mb + r) * KPAD + E_ + ncol[nt]];
#pragma unroll
    for (int nt = 0; nt < 4; ++nt)
#pragma unroll
      for (int r = 0; r < 8; ++r)
        xh[(mb + r) * KPAD + E_ + ncol[nt]] = (_Float16)(rv[nt][r] * hcur[nt][r]);

    // ---- candidate hbar = tanh([x, r*h] @ Wh^T + bh); h_t update ----
    v16h a2r = loadA(2), a3r = loadA(3);
#pragma unroll
    for (int nt = 0; nt < 4; ++nt) {
      v8f acc = {};
      acc = WMMA_F16(afr[0], loadB(sWh, KPAD, ncol[nt], 0), acc);
      acc = WMMA_F16(afr[1], loadB(sWh, KPAD, ncol[nt], 1), acc);
      acc = WMMA_F16(a2r,    loadB(sWh, KPAD, ncol[nt], 2), acc);
      acc = WMMA_F16(a3r,    loadB(sWh, KPAD, ncol[nt], 3), acc);
#pragma unroll
      for (int r = 0; r < 8; ++r) {
        float hb = tanhf_(acc[r] + bhv[nt]);
        float z  = zf[nt][r];
        float ht = (1.0f - z) * hcur[nt][r] + z * hb;
        xh[(mb + r) * KPAD + E_ + ncol[nt]] = (_Float16)ht;   // h for step t+1
      }
    }

    // ---- fused LM head: logits_t = h_t @ Wlm^T + blm ----
    v16h ah0 = loadA(2), ah1 = loadA(3);    // hidden occupies cols 64..127
#pragma unroll
    for (int vt = 0; vt < 2; ++vt) {
      v8f acc = {};
      acc = WMMA_F16(ah0, wlmFrag[vt][0], acc);
      acc = WMMA_F16(ah1, wlmFrag[vt][1], acc);
      if (vcol[vt] < V_) {
#pragma unroll
        for (int r = 0; r < 8; ++r) {
          int b = row0 + mb + r;
          out[(b * T_ + t) * V_ + vcol[vt]] = acc[r] + blmv[vt];
        }
      }
    }
  }
}

extern "C" void kernel_launch(void* const* d_in, const int* in_sizes, int n_in,
                              void* d_out, int out_size, void* d_ws, size_t ws_size,
                              hipStream_t stream) {
  const int*   idx   = (const int*)d_in[0];
  const float* wte   = (const float*)d_in[1];
  const float* start = (const float*)d_in[2];
  const float* Wz    = (const float*)d_in[3];
  const float* bz    = (const float*)d_in[4];
  const float* Wr    = (const float*)d_in[5];
  const float* br    = (const float*)d_in[6];
  const float* Wh    = (const float*)d_in[7];
  const float* bh    = (const float*)d_in[8];
  const float* Wlm   = (const float*)d_in[9];
  const float* blm   = (const float*)d_in[10];
  float* out = (float*)d_out;

  dim3 grid(B_ / (WAVES * RPW));   // 128 blocks
  dim3 block(32 * WAVES);          // 256 threads = 8 waves
  hipLaunchKernelGGL(gru_wmma_kernel, grid, block, 0, stream,
                     idx, wte, start, Wz, bz, Wr, br, Wh, bh, Wlm, blm, out);
}
